// RANNet_25357486916085
// MI455X (gfx1250) — matrix operand order repaired and checked by reading, not compiled, not run on hardware
//
#include <hip/hip_runtime.h>
#include <math.h>

// ---------------------------------------------------------------------------
// RAN forward for MI455X (gfx1250, wave32, WMMA).
// All GEMMs run through v_wmma_f32_16x16x32_bf16 with f32 accumulation.
// ---------------------------------------------------------------------------

typedef __bf16 bf16;
typedef __attribute__((ext_vector_type(16))) __bf16 v16bf;
typedef __attribute__((ext_vector_type(8)))  __bf16 v8bf;
typedef __attribute__((ext_vector_type(4)))  __bf16 v4bf;
typedef __attribute__((ext_vector_type(8)))  float  v8f;

#define BB 64
#define SS 512
#define DD 1024
#define HH 1024
#define CC 1000
#define NPADC 1008   // 63 tiles of 16 cover 1000 cols

// ---- WMMA fragment helpers (layouts per CDNA5 ISA §7.12.2) -----

// A-matrix 16x32 bf16: lane l holds row (l&15); K chunks of 8 at
// k0 + 8*(l>>4) and k0 + 8*(l>>4) + 16.
__device__ __forceinline__ v16bf load_fragA(const bf16* __restrict__ base,
                                            int ld, int row0, int k0, int lane) {
  const int r    = row0 + (lane & 15);
  const int koff = k0 + ((lane >> 4) << 3);
  const bf16* p  = base + (size_t)r * ld + koff;
  v16bf f;
  ((v8bf*)&f)[0] = *(const v8bf*)(p);
  ((v8bf*)&f)[1] = *(const v8bf*)(p + 16);
  return f;
}

// B-matrix 32x16 bf16 sourced from an N x K row-major weight: lane l holds
// column (l&15); K chunks of 8 at k0 + 8*(l>>4) and k0 + 8*(l>>4) + 16.
__device__ __forceinline__ v16bf load_fragB(const bf16* __restrict__ base,
                                            int ld, int col0, int k0, int lane) {
  const int c    = col0 + (lane & 15);
  const int koff = k0 + ((lane >> 4) << 3);
  const bf16* p  = base + (size_t)c * ld + koff;
  v16bf f;
  ((v8bf*)&f)[0] = *(const v8bf*)(p);
  ((v8bf*)&f)[1] = *(const v8bf*)(p + 16);
  return f;
}

__device__ __forceinline__ v8f wmma_bf16(v16bf a, v16bf b, v8f c) {
  return __builtin_amdgcn_wmma_f32_16x16x32_bf16(
      /*neg_a=*/false, a, /*neg_b=*/false, b,
      /*c_mod=*/(short)0, c, /*reuse_a=*/false, /*reuse_b=*/false);
}

__device__ __forceinline__ float sigmoidf_fast(float x) {
  return 1.0f / (1.0f + __expf(-x));
}

// ---------------------------------------------------------------------------
// 1) f32 -> bf16 convert (weights)
// ---------------------------------------------------------------------------
__global__ void cvt_f32_bf16(const float* __restrict__ src,
                             bf16* __restrict__ dst, int n) {
  int i = blockIdx.x * blockDim.x + threadIdx.x;
  if (i < n) dst[i] = (bf16)src[i];
}

// fc_w [1000,1024] -> padded [1024,1024] bf16 (rows >= 1000 zeroed)
__global__ void cvt_fc_padded(const float* __restrict__ src,
                              bf16* __restrict__ dst) {
  int i = blockIdx.x * blockDim.x + threadIdx.x;  // 0 .. 1024*1024-1
  int r = i >> 10;
  dst[i] = (r < CC) ? (bf16)src[(size_t)r * DD + (i & 1023)] : (bf16)0.0f;
}

// ---------------------------------------------------------------------------
// 2) embedding gather: xe[b*S+s, :] = bf16(emb[x[b,s], :])
//    grid = B*S blocks, 256 threads; float4 loads, v4bf stores.
// ---------------------------------------------------------------------------
__global__ void embed_kernel(const int* __restrict__ x,
                             const float* __restrict__ emb,
                             bf16* __restrict__ xe) {
  const int row = blockIdx.x;
  const int tok = x[row];
  const float4* src = (const float4*)(emb + (size_t)tok * DD);
  bf16* dst = xe + (size_t)row * DD;
  float4 v = src[threadIdx.x];
  v4bf o;
  o.x = (bf16)v.x; o.y = (bf16)v.y; o.z = (bf16)v.z; o.w = (bf16)v.w;
  *(v4bf*)(dst + threadIdx.x * 4) = o;
}

// ---------------------------------------------------------------------------
// 3) batched input projections: xf/xi[m, n] = sum_k xe[m,k]*W[n,k] + b[n]
//    M=32768, N=K=1024.  WG = 256 threads (8 waves), 128x128 tile.
//    Waves arranged 4(M) x 2(N): each wave 32x64 = 2x4 fragments.
//    blockIdx.z selects (Wf, Wf_b, xf) vs (Wi, Wi_b, xi).
//    Only xe streams from HBM (64 MB read once); weights live in L2, so we
//    only prefetch-ahead on the A operand — unconditionally, branch-free.
// ---------------------------------------------------------------------------
__global__ void proj_kernel(const bf16* __restrict__ xe,
                            const bf16* __restrict__ Wf,
                            const bf16* __restrict__ Wi,
                            const float* __restrict__ bWf,
                            const float* __restrict__ bWi,
                            bf16* __restrict__ xf,
                            bf16* __restrict__ xi) {
  const bf16*  W    = blockIdx.z ? Wi  : Wf;
  const float* bias = blockIdx.z ? bWi : bWf;
  bf16*        out  = blockIdx.z ? xi  : xf;

  const int tid  = threadIdx.x;
  const int wave = tid >> 5;
  const int lane = tid & 31;
  const int m0 = blockIdx.x * 128 + (wave & 3) * 32;
  const int n0 = blockIdx.y * 128 + (wave >> 2) * 64;

  v8f acc[2][4] = {};

  for (int k = 0; k < DD; k += 32) {
    // prefetch-ahead on the streaming A operand (global_prefetch_b8);
    // address stays inside the xe buffer, speculative misses are dropped.
    __builtin_prefetch(xe + (size_t)m0 * DD + k + 128, 0, 3);

    v16bf a0 = load_fragA(xe, DD, m0,      k, lane);
    v16bf a1 = load_fragA(xe, DD, m0 + 16, k, lane);
    v16bf b0 = load_fragB(W,  DD, n0,      k, lane);
    v16bf b1 = load_fragB(W,  DD, n0 + 16, k, lane);
    v16bf b2 = load_fragB(W,  DD, n0 + 32, k, lane);
    v16bf b3 = load_fragB(W,  DD, n0 + 48, k, lane);
    acc[0][0] = wmma_bf16(a0, b0, acc[0][0]);
    acc[1][0] = wmma_bf16(a1, b0, acc[1][0]);
    acc[0][1] = wmma_bf16(a0, b1, acc[0][1]);
    acc[1][1] = wmma_bf16(a1, b1, acc[1][1]);
    acc[0][2] = wmma_bf16(a0, b2, acc[0][2]);
    acc[1][2] = wmma_bf16(a1, b2, acc[1][2]);
    acc[0][3] = wmma_bf16(a0, b3, acc[0][3]);
    acc[1][3] = wmma_bf16(a1, b3, acc[1][3]);
  }

  // Epilogue: C element r <-> (M = r + 8*(lane>>4), N = lane&15)
  const int mloc = ((lane >> 4) << 3);
  const int nloc = (lane & 15);
#pragma unroll
  for (int ni = 0; ni < 4; ++ni) {
    const int col = n0 + ni * 16 + nloc;
    const float bv = bias[col];
#pragma unroll
    for (int mi = 0; mi < 2; ++mi) {
#pragma unroll
      for (int r = 0; r < 8; ++r) {
        const int row = m0 + mi * 16 + r + mloc;
        out[(size_t)row * HH + col] = (bf16)(acc[mi][ni][r] + bv);
      }
    }
  }
}

// ---------------------------------------------------------------------------
// 4) zero-initialize h state (f32 + bf16 mirror)
// ---------------------------------------------------------------------------
__global__ void init_h(float* __restrict__ hf, bf16* __restrict__ hb) {
  int i = blockIdx.x * blockDim.x + threadIdx.x;  // 0 .. 64*1024-1
  hf[i] = 0.0f;
  hb[i] = (bf16)0.0f;
}

// ---------------------------------------------------------------------------
// 5) recurrence step t (launched 512x, sequential on stream):
//    gF = h @ Ufᵀ, gI = h @ Uiᵀ   (M=64, N=1024, K=1024)
//    f = sigmoid(xf_t + gF); i = sigmoid(xi_t + gI); h' = f*h + i*xe_t
//    grid = 64 WGs (one 16-col tile each), 256 threads = 8 waves.
//    Waves 0-3: Uf with K-quarter (wave&3)*256; waves 4-7: Ui likewise.
//    Partials reduced through 32 KB LDS, then fused elementwise epilogue.
//    Uf/Ui (4 MB bf16 total) stay resident in the 192 MB L2 across steps.
// ---------------------------------------------------------------------------
__global__ void step_kernel(const float* __restrict__ h_in,
                            const bf16* __restrict__ h_in_b,
                            float* __restrict__ h_out,
                            bf16* __restrict__ h_out_b,
                            const bf16* __restrict__ Uf,
                            const bf16* __restrict__ Ui,
                            const bf16* __restrict__ xf,
                            const bf16* __restrict__ xi,
                            const bf16* __restrict__ xe,
                            int t) {
  __shared__ float red[8 * 1024];

  const int tid  = threadIdx.x;
  const int wave = tid >> 5;
  const int lane = tid & 31;
  const int n0   = blockIdx.x * 16;

  const bf16* U = (wave < 4) ? Uf : Ui;
  const int  kq = (wave & 3) * 256;

  v8f acc[4] = {};
  for (int kk = 0; kk < 256; kk += 32) {
    const int k = kq + kk;
    v16bf b = load_fragB(U, HH, n0, k, lane);
#pragma unroll
    for (int mi = 0; mi < 4; ++mi) {
      v16bf a = load_fragA(h_in_b, HH, mi * 16, k, lane);
      acc[mi] = wmma_bf16(a, b, acc[mi]);
    }
  }

  // stash partial 64x16 tile: red[wave][m*16 + n]
  float* myred = red + wave * 1024;
  const int mloc = ((lane >> 4) << 3);
  const int nloc = (lane & 15);
#pragma unroll
  for (int mi = 0; mi < 4; ++mi)
#pragma unroll
    for (int r = 0; r < 8; ++r)
      myred[(mi * 16 + r + mloc) * 16 + nloc] = acc[mi][r];

  __syncthreads();

  for (int e = tid; e < 1024; e += 256) {
    float sF = red[e] + red[1024 + e] + red[2048 + e] + red[3072 + e];
    float sI = red[4096 + e] + red[5120 + e] + red[6144 + e] + red[7168 + e];
    const int m   = e >> 4;
    const int col = n0 + (e & 15);
    const size_t tix = ((size_t)m * SS + t) * HH + col;
    const float fv = sigmoidf_fast((float)xf[tix] + sF);
    const float iv = sigmoidf_fast((float)xi[tix] + sI);
    const float hp = h_in[m * HH + col];
    const float hn = fv * hp + iv * (float)xe[tix];
    h_out[m * HH + col]   = hn;
    h_out_b[m * HH + col] = (bf16)hn;
  }
}

// ---------------------------------------------------------------------------
// 6) final FC: out[64,1000] = h_last @ fc_wᵀ + fc_b (fc padded to 1024 cols)
//    grid = 63 WGs (16-col tiles), 8 waves split K (128 each), LDS reduce.
// ---------------------------------------------------------------------------
__global__ void fc_kernel(const bf16* __restrict__ h_b,
                          const bf16* __restrict__ W,
                          const float* __restrict__ bias,
                          float* __restrict__ out) {
  __shared__ float red[8 * 1024];

  const int tid  = threadIdx.x;
  const int wave = tid >> 5;
  const int lane = tid & 31;
  const int n0   = blockIdx.x * 16;
  const int kq   = wave * 128;

  v8f acc[4] = {};
  for (int kk = 0; kk < 128; kk += 32) {
    const int k = kq + kk;
    v16bf b = load_fragB(W, DD, n0, k, lane);
#pragma unroll
    for (int mi = 0; mi < 4; ++mi) {
      v16bf a = load_fragA(h_b, HH, mi * 16, k, lane);
      acc[mi] = wmma_bf16(a, b, acc[mi]);
    }
  }

  float* myred = red + wave * 1024;
  const int mloc = ((lane >> 4) << 3);
  const int nloc = (lane & 15);
#pragma unroll
  for (int mi = 0; mi < 4; ++mi)
#pragma unroll
    for (int r = 0; r < 8; ++r)
      myred[(mi * 16 + r + mloc) * 16 + nloc] = acc[mi][r];

  __syncthreads();

  for (int e = tid; e < 1024; e += 256) {
    float s = red[e] + red[1024 + e] + red[2048 + e] + red[3072 + e] +
              red[4096 + e] + red[5120 + e] + red[6144 + e] + red[7168 + e];
    const int m   = e >> 4;
    const int col = n0 + (e & 15);
    if (col < CC) out[(size_t)m * CC + col] = s + bias[col];
  }
}

// ---------------------------------------------------------------------------
// host launcher
// ---------------------------------------------------------------------------
extern "C" void kernel_launch(void* const* d_in, const int* in_sizes, int n_in,
                              void* d_out, int out_size, void* d_ws, size_t ws_size,
                              hipStream_t stream) {
  (void)in_sizes; (void)n_in; (void)out_size; (void)ws_size;

  const int*   x    = (const int*)  d_in[0];
  const float* emb  = (const float*)d_in[1];
  const float* Wf_w = (const float*)d_in[2];
  const float* Wf_b = (const float*)d_in[3];
  const float* Uf_w = (const float*)d_in[4];
  const float* Wi_w = (const float*)d_in[5];
  const float* Wi_b = (const float*)d_in[6];
  const float* Ui_w = (const float*)d_in[7];
  const float* fc_w = (const float*)d_in[8];
  const float* fc_b = (const float*)d_in[9];
  float* out = (float*)d_out;

  char* ws = (char*)d_ws;
  size_t off = 0;
  auto alloc = [&](size_t bytes) -> void* {
    void* p = ws + off;
    off += (bytes + 255) & ~(size_t)255;
    return p;
  };

  const size_t MROWS = (size_t)BB * SS;            // 32768
  bf16* xe  = (bf16*)alloc(MROWS * DD * sizeof(bf16));     // 64 MB
  bf16* WfB = (bf16*)alloc((size_t)HH * DD * sizeof(bf16));
  bf16* WiB = (bf16*)alloc((size_t)HH * DD * sizeof(bf16));
  bf16* UfB = (bf16*)alloc((size_t)HH * HH * sizeof(bf16));
  bf16* UiB = (bf16*)alloc((size_t)HH * HH * sizeof(bf16));
  bf16* fcB = (bf16*)alloc((size_t)1024 * DD * sizeof(bf16));
  bf16* xf  = (bf16*)alloc(MROWS * HH * sizeof(bf16));     // 64 MB
  bf16* xi  = (bf16*)alloc(MROWS * HH * sizeof(bf16));     // 64 MB
  float* hF[2];
  bf16*  hB[2];
  hF[0] = (float*)alloc((size_t)BB * HH * sizeof(float));
  hF[1] = (float*)alloc((size_t)BB * HH * sizeof(float));
  hB[0] = (bf16*) alloc((size_t)BB * HH * sizeof(bf16));
  hB[1] = (bf16*) alloc((size_t)BB * HH * sizeof(bf16));

  // weight conversion to bf16
  const int NW = HH * DD;  // 1048576
  cvt_f32_bf16<<<NW / 256, 256, 0, stream>>>(Wf_w, WfB, NW);
  cvt_f32_bf16<<<NW / 256, 256, 0, stream>>>(Wi_w, WiB, NW);
  cvt_f32_bf16<<<NW / 256, 256, 0, stream>>>(Uf_w, UfB, NW);
  cvt_f32_bf16<<<NW / 256, 256, 0, stream>>>(Ui_w, UiB, NW);
  cvt_fc_padded<<<NW / 256, 256, 0, stream>>>(fc_w, fcB);

  // embedding gather
  embed_kernel<<<(int)MROWS, 256, 0, stream>>>(x, emb, xe);

  // batched input projections (xf, xi)
  proj_kernel<<<dim3(MROWS / 128, HH / 128, 2), 256, 0, stream>>>(
      xe, WfB, WiB, Wf_b, Wi_b, xf, xi);

  // h0 = 0
  init_h<<<(BB * HH) / 256, 256, 0, stream>>>(hF[0], hB[0]);

  // sequential recurrence: step t reads buf[t&1], writes buf[(t+1)&1]
  for (int t = 0; t < SS; ++t) {
    const int in_i  = t & 1;
    const int out_i = in_i ^ 1;
    step_kernel<<<HH / 16, 256, 0, stream>>>(
        hF[in_i], hB[in_i], hF[out_i], hB[out_i],
        UfB, UiB, xf, xi, xe, t);
  }
  // after 512 steps final state is in buf[0]

  // final FC
  fc_kernel<<<NPADC / 16, 256, 0, stream>>>(hB[0], fcB, fc_b, out);
}
